// RotationAttentionFusion_58265526337587
// MI455X (gfx1250) — matrix operand order repaired and checked by reading, not compiled, not run on hardware
//
#include <hip/hip_runtime.h>

typedef __bf16 bf16;
typedef __attribute__((ext_vector_type(8)))  __bf16 bf16x8;
typedef __attribute__((ext_vector_type(16))) __bf16 bf16x16;
typedef __attribute__((ext_vector_type(8)))  float  v8f;
typedef int v4i_ __attribute__((vector_size(16)));   // matches builtin pointee

#define BATCH 4
#define CQK   64
#define CR    256
#define LSEQ  8192
#define KPAD  72   // halves per row (144 B): bank-conflict-free b128 row access

__device__ __forceinline__ v8f wmma_bf16(bf16x16 a, bf16x16 b, v8f c) {
  return __builtin_amdgcn_wmma_f32_16x16x32_bf16(false, a, false, b, (short)0, c, false, false);
}

__device__ __forceinline__ bf16x16 join8(bf16x8 lo, bf16x8 hi) {
  bf16x16 r;
#pragma unroll
  for (int i = 0; i < 8; ++i) { r[i] = lo[i]; r[i + 8] = hi[i]; }
  return r;
}

// 16B global -> LDS async copy (no VGPR round trip, tracked by ASYNCcnt)
__device__ __forceinline__ void cp16_async(void* lds, const void* g) {
  __builtin_amdgcn_global_load_async_to_lds_b128(
      (__attribute__((address_space(1))) v4i_*)g,
      (__attribute__((address_space(3))) v4i_*)lds, 0, 0);
}
__device__ __forceinline__ void wait_async0() {
  asm volatile("s_wait_asynccnt 0x0" ::: "memory");
}

// ---------------------------------------------------------------------------
// Kernel 1: input projections. q,k -> bf16 [b][l][64]; v(+pos) -> bf16 [b][c][l]
// ---------------------------------------------------------------------------
__global__ __launch_bounds__(256) void proj_kernel(
    const float* __restrict__ x,    // [B,256,L]
    const float* __restrict__ Wq, const float* __restrict__ bq,
    const float* __restrict__ Wk, const float* __restrict__ bk,
    const float* __restrict__ Wv, const float* __restrict__ bv,
    const float* __restrict__ pos, // [1,256,L]
    bf16* __restrict__ qh, bf16* __restrict__ kh, bf16* __restrict__ vh) {
  __shared__ float xs[CR * 64];   // [i][l] 64 KB
  const int tid = threadIdx.x;
  const int l0  = blockIdx.x * 64;
  const int b   = blockIdx.y;

  // stage x tile via async DMA: rows i contiguous in l
#pragma unroll
  for (int kk = 0; kk < 16; ++kk) {
    int f = tid + kk * 256;           // 4096 chunks of 16B
    int row = f >> 4, cc = f & 15;
    cp16_async((char*)xs + (size_t)f * 16,
               x + ((size_t)b * CR + row) * LSEQ + l0 + cc * 4);
  }
  wait_async0();
  __syncthreads();

  const int l = tid & 63;
  const int g = tid >> 6;           // 0..3

  // q and k: 16 output channels per group
  for (int j = 0; j < 16; ++j) {
    int o = g * 16 + j;
    float aq = bq[o], ak = bk[o];
    const float* wq = Wq + (size_t)o * CR;
    const float* wk = Wk + (size_t)o * CR;
#pragma unroll 8
    for (int i = 0; i < CR; ++i) {
      float xv = xs[i * 64 + l];
      aq += wq[i] * xv;
      ak += wk[i] * xv;
    }
    size_t ridx = ((size_t)b * LSEQ + l0 + l) * CQK + o;
    qh[ridx] = (bf16)(aq * 0.125f);   // fold scale = C^-0.5
    kh[ridx] = (bf16)ak;
  }
  // v: 64 output channels per group, c-major layout + pos_emb
  for (int j = 0; j < 64; ++j) {
    int o = g * 64 + j;
    float av = bv[o] + pos[(size_t)o * LSEQ + l0 + l];
    const float* wv = Wv + (size_t)o * CR;
#pragma unroll 8
    for (int i = 0; i < CR; ++i) av += wv[i] * xs[i * 64 + l];
    vh[((size_t)b * CR + o) * LSEQ + l0 + l] = (bf16)av;
  }
}

// ---------------------------------------------------------------------------
// Kernel 2: flash attention. 8 waves/block, 16 query rows per wave (BM=128),
// KV tiles of 64, double-buffered via async global->LDS DMA.
// ---------------------------------------------------------------------------
__device__ __forceinline__ void issue_tiles(
    const bf16* __restrict__ kh, const bf16* __restrict__ vh,
    bf16* dK, bf16* dV, int b, int m0, int tid) {
  // K tile: 64 rows x 128B, source is one contiguous 8KB slab
  const char* srcK = (const char*)(kh + ((size_t)b * LSEQ + m0) * CQK);
#pragma unroll
  for (int i = 0; i < 2; ++i) {
    int f = tid + i * 256;            // 512 chunks
    int row = f >> 3, cc = f & 7;
    cp16_async((char*)dK + row * (KPAD * 2) + cc * 16, srcK + (size_t)f * 16);
  }
  // V tile: 256 rows x 128B, rows strided by L in global
#pragma unroll
  for (int i = 0; i < 8; ++i) {
    int f = tid + i * 256;            // 2048 chunks
    int row = f >> 3, cc = f & 7;
    cp16_async((char*)dV + row * (KPAD * 2) + cc * 16,
               (const char*)(vh + ((size_t)b * CR + row) * LSEQ + m0) + cc * 16);
  }
}

__global__ __launch_bounds__(256) void attn_kernel(
    const bf16* __restrict__ qh, const bf16* __restrict__ kh,
    const bf16* __restrict__ vh, float* __restrict__ ao /* [b][l][256] */) {
  __shared__ bf16 ldsK[2][64 * KPAD];      // 2 x 9  KB  [m][c]
  __shared__ bf16 ldsV[2][CR * KPAD];      // 2 x 36 KB  [c][m]
  __shared__ bf16 ldsP[8 * 16 * KPAD];     // 18 KB per-wave P scratch [row][m]

  const int tid  = threadIdx.x;
  const int lane = tid & 31;
  const int wave = tid >> 5;
  const int half = lane >> 4;           // 0 | 1
  const int nn   = lane & 15;
  const int b    = blockIdx.y;
  const int l0   = blockIdx.x * 128;

  // ---- this wave's 16 query rows in WMMA A-layout ----
  const bf16* qrow = qh + ((size_t)b * LSEQ + l0 + wave * 16 + nn) * CQK;
  bf16x16 qa0 = join8(*(const bf16x8*)(qrow + half * 8),
                      *(const bf16x8*)(qrow + 16 + half * 8));
  bf16x16 qa1 = join8(*(const bf16x8*)(qrow + 32 + half * 8),
                      *(const bf16x8*)(qrow + 48 + half * 8));

  v8f acc[16];
#pragma unroll
  for (int ct = 0; ct < 16; ++ct) acc[ct] = (v8f){};
  float mrow[8], lrow[8];
#pragma unroll
  for (int r = 0; r < 8; ++r) { mrow[r] = -1.0e30f; lrow[r] = 0.0f; }

  issue_tiles(kh, vh, ldsK[0], ldsV[0], b, 0, tid);   // prologue DMA

  int buf = 0;
  for (int m0 = 0; m0 < LSEQ; m0 += 64) {
    wait_async0();        // my DMA chunks for 'buf' done
    __syncthreads();      // everyone's done
    if (m0 + 64 < LSEQ)   // overlap next tile's DMA with this tile's math
      issue_tiles(kh, vh, ldsK[buf ^ 1], ldsV[buf ^ 1], b, m0 + 64, tid);

    const bf16* Kt = ldsK[buf];
    const bf16* Vt = ldsV[buf];

    // ---- S = Q K^T : 4 tiles of 16 key columns ----
    v8f s[4];
#pragma unroll
    for (int nt = 0; nt < 4; ++nt) {
      const bf16* kr = Kt + (nt * 16 + nn) * KPAD + half * 8;
      bf16x16 b0 = join8(*(const bf16x8*)(kr),      *(const bf16x8*)(kr + 16));
      bf16x16 b1 = join8(*(const bf16x8*)(kr + 32), *(const bf16x8*)(kr + 48));
      v8f z = (v8f){};
      s[nt] = wmma_bf16(qa0, b0, z);
      s[nt] = wmma_bf16(qa1, b1, s[nt]);
    }

    // ---- online softmax per row (row = r + 8*half, cols across 16 lanes) ----
    float corr[8];
#pragma unroll
    for (int r = 0; r < 8; ++r) {
      float sm = fmaxf(fmaxf(s[0][r], s[1][r]), fmaxf(s[2][r], s[3][r]));
#pragma unroll
      for (int msk = 1; msk < 16; msk <<= 1) sm = fmaxf(sm, __shfl_xor(sm, msk, 32));
      float newm = fmaxf(mrow[r], sm);
      corr[r] = __expf(mrow[r] - newm);
      float psum = 0.0f;
#pragma unroll
      for (int nt = 0; nt < 4; ++nt) {
        float pe = __expf(s[nt][r] - newm);
        psum += pe;
        ldsP[wave * (16 * KPAD) + (r + 8 * half) * KPAD + nt * 16 + nn] = (bf16)pe;
      }
#pragma unroll
      for (int msk = 1; msk < 16; msk <<= 1) psum += __shfl_xor(psum, msk, 32);
      lrow[r] = lrow[r] * corr[r] + psum;
      mrow[r] = newm;
    }
#pragma unroll
    for (int ct = 0; ct < 16; ++ct)
#pragma unroll
      for (int r = 0; r < 8; ++r) acc[ct][r] *= corr[r];

    // ---- O += P V : reload P in A-layout from per-wave scratch ----
    const bf16* pb = ldsP + wave * (16 * KPAD) + nn * KPAD;
    bf16x16 pa0 = join8(*(const bf16x8*)(pb + half * 8),
                        *(const bf16x8*)(pb + 16 + half * 8));
    bf16x16 pa1 = join8(*(const bf16x8*)(pb + 32 + half * 8),
                        *(const bf16x8*)(pb + 48 + half * 8));
#pragma unroll
    for (int ct = 0; ct < 16; ++ct) {
      const bf16* vr = Vt + (ct * 16 + nn) * KPAD + half * 8;
      bf16x16 b0 = join8(*(const bf16x8*)(vr),      *(const bf16x8*)(vr + 16));
      bf16x16 b1 = join8(*(const bf16x8*)(vr + 32), *(const bf16x8*)(vr + 48));
      acc[ct] = wmma_bf16(pa0, b0, acc[ct]);
      acc[ct] = wmma_bf16(pa1, b1, acc[ct]);
    }
    buf ^= 1;
  }

  // ---- normalize and write O (fp32, [b][l][256]) ----
#pragma unroll
  for (int ct = 0; ct < 16; ++ct) {
#pragma unroll
    for (int r = 0; r < 8; ++r) {
      int l = l0 + wave * 16 + r + 8 * half;
      int c = ct * 16 + nn;
      ao[((size_t)b * LSEQ + l) * CR + c] = acc[ct][r] / lrow[r];
    }
  }
}

// ---------------------------------------------------------------------------
// Kernel 3: output projection y[b][o][l] = Wo[o][:] . ao[b][l][:] + bo[o]
// ---------------------------------------------------------------------------
__global__ __launch_bounds__(256) void oproj_kernel(
    const float* __restrict__ ao, const float* __restrict__ Wo,
    const float* __restrict__ bo, float* __restrict__ out) {
  __shared__ float ot[64 * CR];   // 64 KB, contiguous slab of ao
  const int tid = threadIdx.x;
  const int l0  = blockIdx.x * 64;
  const int b   = blockIdx.y;

  const char* src = (const char*)(ao + ((size_t)b * LSEQ + l0) * CR);
#pragma unroll
  for (int kk = 0; kk < 16; ++kk) {
    int f = tid + kk * 256;
    cp16_async((char*)ot + (size_t)f * 16, src + (size_t)f * 16);
  }
  wait_async0();
  __syncthreads();

  const int o  = tid & 63;
  const int lg = tid >> 6;
  const float* wo = Wo + (size_t)o * CR;
  const float  bb = bo[o];
  for (int j = 0; j < 16; ++j) {
    int l = lg + 4 * j;
    float acc = bb;
#pragma unroll 8
    for (int c = 0; c < CR; ++c) acc += wo[c] * ot[l * CR + c];
    out[((size_t)b * CQK + o) * LSEQ + l0 + l] = acc;
  }
}

// ---------------------------------------------------------------------------
extern "C" void kernel_launch(void* const* d_in, const int* in_sizes, int n_in,
                              void* d_out, int out_size, void* d_ws, size_t ws_size,
                              hipStream_t stream) {
  const float* x   = (const float*)d_in[0];
  const float* Wq  = (const float*)d_in[1];
  const float* bq  = (const float*)d_in[2];
  const float* Wk  = (const float*)d_in[3];
  const float* bk  = (const float*)d_in[4];
  const float* Wv  = (const float*)d_in[5];
  const float* bv  = (const float*)d_in[6];
  const float* Wo  = (const float*)d_in[7];
  const float* bo  = (const float*)d_in[8];
  const float* pos = (const float*)d_in[9];
  float* out = (float*)d_out;

  char* ws = (char*)d_ws;
  bf16*  qh = (bf16*)(ws);                                    // 4 MB
  bf16*  kh = (bf16*)(ws + (size_t)4  * 1024 * 1024);         // 4 MB
  bf16*  vh = (bf16*)(ws + (size_t)8  * 1024 * 1024);         // 16 MB
  float* ao = (float*)(ws + (size_t)24 * 1024 * 1024);        // 32 MB

  proj_kernel<<<dim3(LSEQ / 64, BATCH), 256, 0, stream>>>(
      x, Wq, bq, Wk, bk, Wv, bv, pos, qh, kh, vh);
  attn_kernel<<<dim3(LSEQ / 128, BATCH), 256, 0, stream>>>(qh, kh, vh, ao);
  oproj_kernel<<<dim3(LSEQ / 64, BATCH), 256, 0, stream>>>(ao, Wo, bo, out);
}
